// XCubeVAE_63496796504730
// MI455X (gfx1250) — matrix operand order repaired and checked by reading, not compiled, not run on hardware
//
#include <hip/hip_runtime.h>

// ---------------------------------------------------------------------------
// XCubeVAE dense U-Net on gfx1250.
// Implicit-GEMM conv3d via v_wmma_f32_16x16x32_bf16, f32 accumulate.
//   - zero-page halo handling: no EXEC predication anywhere in the conv kernel
//   - 16x64 output per wave (4 WMMA accumulators, A-fragment reused 4x)
//   - all dims are powers of two -> shift/mask addressing only
// ---------------------------------------------------------------------------

typedef __attribute__((ext_vector_type(16))) __bf16 v16bf;
typedef __attribute__((ext_vector_type(8)))  float  v8f;

union Frag32 { uint4 u[2]; v16bf v; };   // 32 B = 8 VGPRs = one 16-bit WMMA operand

__device__ __forceinline__ unsigned short f2bf(float f) {
  unsigned int b = __builtin_bit_cast(unsigned int, f);
  unsigned int r = (b + 0x7FFFu + ((b >> 16) & 1u)) >> 16;   // RNE
  return (unsigned short)r;
}

// fp32 NCDHW activation -> bf16 voxel-major [S][C] (channel-contiguous)
__global__ void k_cvt_dhwc_bf16(const float* __restrict__ src,
                                unsigned short* __restrict__ dst,
                                int n, int lc, int ls3) {   // lc=log2(C), ls3=log2(S)
  int idx = blockIdx.x * 256 + threadIdx.x;
  if (idx >= n) return;
  int c = idx & ((1 << lc) - 1);
  int v = idx >> lc;
  dst[idx] = f2bf(src[((long)c << ls3) + v]);
}

// fp32 OIDHW weights -> bf16 [tap][Cout][Cin]
__global__ void k_pack_w(const float* __restrict__ w,
                         unsigned short* __restrict__ wpk,
                         int n, int lci, int lco, int KT) {
  int idx = blockIdx.x * 256 + threadIdx.x;
  if (idx >= n) return;
  int cin  = idx & ((1 << lci) - 1);
  int r    = idx >> lci;
  int cout = r & ((1 << lco) - 1);
  int tap  = r >> lco;
  wpk[idx] = f2bf(w[(((long)cout << lci) + cin) * KT + tap]);
}

// 512-byte zero page for padded-halo B fragments
__global__ void k_zero_page(unsigned short* __restrict__ p) {
  p[threadIdx.x] = 0;
}

// ---------------------------------------------------------------------------
// conv3d SAME, bf16 WMMA:  M = Cout, N = S voxels, K = Cin * ktaps
// block = 8 waves, tile 32(M) x 256(N); each wave: 16(M) x 64(N) = 4 D-tiles.
// ---------------------------------------------------------------------------
__global__ __launch_bounds__(256)
void k_conv3d_wmma(const unsigned short* __restrict__ act,  // bf16 [S][Cin] (+zero page)
                   const unsigned short* __restrict__ wpk,  // bf16 [KT][Cout][Cin]
                   const float* __restrict__ bias,          // [Cout]
                   float* __restrict__ out,                 // [Cout][S]
                   int ld, int lci, int Cin, int ktaps, int relu, long zoff) {
  const int D    = 1 << ld;
  const int S    = 1 << (3 * ld);
  const int tid  = threadIdx.x;
  const int wave = tid >> 5;
  const int lane = tid & 31;
  const int half = lane >> 4;      // 0: lanes 0-15, 1: lanes 16-31
  const int l    = lane & 15;

  const int m_sub = wave & 1;      // 2 wave-rows (M)
  const int n_sub = wave >> 1;     // 4 wave-cols (N)
  const int cout0 = blockIdx.y * 32 + m_sub * 16;
  const int vox0  = blockIdx.x * 256 + n_sub * 64;

  // decode the 4 B-column voxels owned by this lane (N = l in each 16-wide tile)
  int xs[4], ys[4], zs[4];
#pragma unroll
  for (int nt = 0; nt < 4; ++nt) {
    int v = vox0 + nt * 16 + l;
    xs[nt] = v & (D - 1);
    ys[nt] = (v >> ld) & (D - 1);
    zs[nt] = v >> (2 * ld);
  }

  v8f acc[4] = {{}, {}, {}, {}};

  for (int tap = 0; tap < ktaps; ++tap) {
    int dz = 0, dy = 0, dx = 0;
    if (ktaps == 27) { dz = tap / 9 - 1; dy = (tap / 3) % 3 - 1; dx = tap % 3 - 1; }

    long bb[4];
#pragma unroll
    for (int nt = 0; nt < 4; ++nt) {
      const int zz = zs[nt] + dz, yy = ys[nt] + dy, xx = xs[nt] + dx;
      const bool valid = ((unsigned)zz < (unsigned)D) &
                         ((unsigned)yy < (unsigned)D) &
                         ((unsigned)xx < (unsigned)D);
      const long vlin = ((long)(((zz << ld) + yy) << ld) + xx) << lci;
      bb[nt] = valid ? vlin : zoff;          // halo lanes read zeros, EXEC untouched
    }
    const long abase = (((long)tap << lci) * (S >= 0 ? 1 : 1)) * 0 +  // (kept simple below)
                       ((((long)tap * (long)(1 << 0))));              // placeholder removed
    // real A row base: [tap][cout0+l][Cin]
    const long arow = (((long)tap * (long)gridDim.y * 32 + (cout0 + l)) << lci);

    for (int c0 = 0; c0 < Cin; c0 += 32) {
      // A fragment: row M = l; half0 -> K 0..15, half1 -> K 16..31
      Frag32 af;
      af.u[0] = *(const uint4*)(wpk + arow + c0 + half * 16);
      af.u[1] = *(const uint4*)(wpk + arow + c0 + half * 16 + 8);

      // 4 B fragments: lane half selects K 0..15 / 16..31 (channels)
      Frag32 b0, b1, b2, b3;
      {
        const uint4* p0 = (const uint4*)(act + bb[0] + c0 + half * 16);
        const uint4* p1 = (const uint4*)(act + bb[1] + c0 + half * 16);
        const uint4* p2 = (const uint4*)(act + bb[2] + c0 + half * 16);
        const uint4* p3 = (const uint4*)(act + bb[3] + c0 + half * 16);
        b0.u[0] = p0[0]; b0.u[1] = p0[1];
        b1.u[0] = p1[0]; b1.u[1] = p1[1];
        b2.u[0] = p2[0]; b2.u[1] = p2[1];
        b3.u[0] = p3[0]; b3.u[1] = p3[1];
      }

      acc[0] = __builtin_amdgcn_wmma_f32_16x16x32_bf16(false, af.v, false, b0.v,
                                                       (short)0, acc[0], false, false);
      acc[1] = __builtin_amdgcn_wmma_f32_16x16x32_bf16(false, af.v, false, b1.v,
                                                       (short)0, acc[1], false, false);
      acc[2] = __builtin_amdgcn_wmma_f32_16x16x32_bf16(false, af.v, false, b2.v,
                                                       (short)0, acc[2], false, false);
      acc[3] = __builtin_amdgcn_wmma_f32_16x16x32_bf16(false, af.v, false, b3.v,
                                                       (short)0, acc[3], false, false);
    }
  }

  // Epilogue. C/D layout: VGPR r, lane half -> M = r + 8*half, N = l.
  float bv[8];
#pragma unroll
  for (int r = 0; r < 8; ++r) bv[r] = bias[cout0 + r + half * 8];

#pragma unroll
  for (int nt = 0; nt < 4; ++nt) {
#pragma unroll
    for (int r = 0; r < 8; ++r) {
      const int m = r + half * 8;
      float val = acc[nt][r] + bv[r];
      if (relu) val = fmaxf(val, 0.0f);
      out[(long)(cout0 + m) * S + vox0 + nt * 16 + l] = val;
    }
  }
}

// 2x2x2 maxpool on fp32 [C][2Do]^3 -> [C][Do]^3   (lo = log2 Do)
__global__ void k_maxpool2(const float* __restrict__ in, float* __restrict__ out,
                           int n, int lo) {
  int idx = blockIdx.x * 256 + threadIdx.x;
  if (idx >= n) return;
  int Do = 1 << lo;
  int v = idx & ((Do * Do * Do) - 1), c = idx >> (3 * lo);
  int x = v & (Do - 1), y = (v >> lo) & (Do - 1), z = v >> (2 * lo);
  int li = lo + 1;
  long base = (long)c << (3 * li);
  float m = -3.402823466e+38f;
#pragma unroll
  for (int k = 0; k < 8; ++k) {
    int zz = 2 * z + (k >> 2), yy = 2 * y + ((k >> 1) & 1), xx = 2 * x + (k & 1);
    m = fmaxf(m, in[base + ((long)(((zz << li) + yy) << li) + xx)]);
  }
  out[idx] = m;
}

// out[C][(2Di)^3] = repeat2( in[C][Di^3] * mask[Di^3] )   (li = log2 Di)
__global__ void k_upmask(const float* __restrict__ in, const int* __restrict__ mask,
                         float* __restrict__ out, int n, int li) {
  int idx = blockIdx.x * 256 + threadIdx.x;
  if (idx >= n) return;
  int lo = li + 1, Do = 1 << lo;
  int v = idx & ((Do * Do * Do) - 1), c = idx >> (3 * lo);
  int x = v & (Do - 1), y = (v >> lo) & (Do - 1), z = v >> (2 * lo);
  int zi = z >> 1, yi = y >> 1, xi = x >> 1;
  int mi = ((zi << li) + yi << li) + xi;
  mi = (((zi << li) + yi) << li) + xi;
  out[idx] = in[((long)c << (3 * li)) + mi] * (float)mask[mi];
}

// ---------------------------------------------------------------------------
// Host orchestration
// ---------------------------------------------------------------------------
static inline int ilog2(int v) { int l = 0; while ((1 << l) < v) ++l; return l; }

extern "C" void kernel_launch(void* const* d_in, const int* in_sizes, int n_in,
                              void* d_out, int out_size, void* d_ws, size_t ws_size,
                              hipStream_t stream) {
  (void)in_sizes; (void)n_in; (void)out_size; (void)ws_size;

  const float* x_in    = (const float*)d_in[0];
  const int*   mask_d1 = (const int*)d_in[2];
  const int*   mask_d2 = (const int*)d_in[3];
  const int*   mask_d3 = (const int*)d_in[4];

  // Workspace carve (max activation: 64ch @ 64^3 = 16,777,216 elems)
  const long MAXE = 64L * 64 * 64 * 64;
  char* ws = (char*)d_ws;
  float*          bufA  = (float*)ws;
  float*          bufB  = (float*)(ws + MAXE * 4);
  unsigned short* actbf = (unsigned short*)(ws + 2 * MAXE * 4);          // MAXE + 256 elems
  unsigned short* wpk   = (unsigned short*)(ws + 2 * MAXE * 4 + (MAXE + 256) * 2);
  const long zoff = MAXE;                                                // zero page offset

  k_zero_page<<<dim3(1), dim3(256), 0, stream>>>(actbf + zoff);

  // Param leaves in setup_inputs() insertion order:
  // pre, enc[0..3][0..3], pre_kl, post_kl, dec[0..2][0..3]; each {w,b}
  int pi = 5;
  auto next_wb = [&](const float** w, const float** b) {
    *w = (const float*)d_in[pi++];
    *b = (const float*)d_in[pi++];
  };

  auto conv = [&](const float* src, float* dst, int D, int Cin, int Cout,
                  int KT, int relu) {
    const int S = D * D * D;
    const int ld = ilog2(D), lci = ilog2(Cin), lco = ilog2(Cout);
    const float *w, *b;
    next_wb(&w, &b);
    int n1 = S * Cin;
    k_cvt_dhwc_bf16<<<dim3((n1 + 255) / 256), dim3(256), 0, stream>>>(
        src, actbf, n1, lci, 3 * ld);
    int n2 = KT * Cout * Cin;
    k_pack_w<<<dim3((n2 + 255) / 256), dim3(256), 0, stream>>>(
        w, wpk, n2, lci, lco, KT);
    k_conv3d_wmma<<<dim3(S / 256, Cout / 32), dim3(256), 0, stream>>>(
        actbf, wpk, b, dst, ld, lci, Cin, KT, relu, zoff);
  };

  static const int enc_ch[4][4][2] = {
      {{32, 32}, {32, 32}, {32, 32}, {32, 32}},
      {{32, 32}, {32, 32}, {32, 32}, {32, 64}},
      {{64, 64}, {64, 64}, {64, 64}, {64, 128}},
      {{128, 128}, {128, 128}, {128, 128}, {128, 256}},
  };
  static const int dec_ch[3][4][2] = {
      {{256, 128}, {128, 256}, {256, 128}, {128, 128}},
      {{128, 64}, {64, 128}, {128, 64}, {64, 64}},
      {{64, 32}, {32, 64}, {64, 32}, {32, 32}},
  };

  float* cur = bufA;
  float* nxt = bufB;
  int D = 64;
  int C = 32;

  // pre-conv 1x1x1, no ReLU
  conv(x_in, cur, D, 32, 32, 1, 0);

  // encoders
  for (int lvl = 0; lvl < 4; ++lvl) {
    if (lvl > 0) {
      int Do = D / 2;
      int n = C * Do * Do * Do;
      k_maxpool2<<<dim3((n + 255) / 256), dim3(256), 0, stream>>>(cur, nxt, n, ilog2(Do));
      D = Do;
      float* t = cur; cur = nxt; nxt = t;
    }
    for (int j = 0; j < 4; ++j) {
      conv(cur, nxt, D, enc_ch[lvl][j][0], enc_ch[lvl][j][1], 27, 1);
      C = enc_ch[lvl][j][1];
      float* t = cur; cur = nxt; nxt = t;
    }
  }

  // pre_kl + post_kl: 8 x (256,256) @ 8^3
  for (int j = 0; j < 8; ++j) {
    conv(cur, nxt, D, 256, 256, 27, 1);
    float* t = cur; cur = nxt; nxt = t;
  }

  // decoder: mask-gated x2 upsample + 4 convs per stage
  const int* masks[3] = {mask_d3, mask_d2, mask_d1};
  for (int st = 0; st < 3; ++st) {
    int n = C * (2 * D) * (2 * D) * (2 * D);
    k_upmask<<<dim3((n + 255) / 256), dim3(256), 0, stream>>>(cur, masks[st], nxt, n, ilog2(D));
    D *= 2;
    float* t = cur; cur = nxt; nxt = t;
    for (int j = 0; j < 4; ++j) {
      const bool last = (st == 2) && (j == 3);
      float* dst = last ? (float*)d_out : nxt;
      conv(cur, dst, D, dec_ch[st][j][0], dec_ch[st][j][1], 27, 1);
      C = dec_ch[st][j][1];
      if (!last) { float* t2 = cur; cur = nxt; nxt = t2; }
    }
  }
}
